// ReformerLayer_43164421325469
// MI455X (gfx1250) — matrix operand order repaired and checked by reading, not compiled, not run on hardware
//
#include <hip/hip_runtime.h>
#include <math.h>

// ---------------------------------------------------------------------------
// Reformer layer on gfx1250 (MI455X).
//  - all matmuls: v_wmma_f32_16x16x32_bf16 (bf16 in, f32 accumulate)
//  - weights pre-converted to bf16 once per call; activations produced in bf16
//  - pure-copy LDS staging (GEMM A tile, attention Q gather) uses
//    global_load_async_to_lds_b128 (+ s_wait_asynccnt), B tiles transposed in
//    LDS so every WMMA fragment is 2x ds_load_b128
//  - global_prefetch for next k-step tiles
// Workspace requirement: ~215 MB.
// ---------------------------------------------------------------------------

constexpr int CBATCH = 2;        // B
constexpr int CS     = 4096;     // seq
constexpr int CDIM   = 1024;     // d_model
constexpr int CHEAD  = 16;       // heads
constexpr int CNBUCK = 64;       // n_buckets
constexpr int CR     = 4;        // n_hashes
constexpr int CDFF   = 4096;     // d_ff
constexpr int CDH    = CDIM / CHEAD;   // 64 head dim
constexpr int CCH    = CS / CNBUCK;    // 64 chunk length
constexpr int CNCH   = CS / CCH;       // 64 chunks
constexpr int NBH    = CNBUCK / 2;     // 32 rotations
constexpr int KE     = 2 * CCH;        // 128 extended keys

typedef __attribute__((ext_vector_type(16))) __bf16 v16bf;
typedef __attribute__((ext_vector_type(8)))  __bf16 bf16x8;
typedef __attribute__((ext_vector_type(4)))  __bf16 bf16x4;
typedef __attribute__((ext_vector_type(8)))  float  v8f;
typedef __attribute__((ext_vector_type(4)))  float  f32x4;

static __device__ __forceinline__ v8f wmma_bf16(const v16bf& a, const v16bf& b, const v8f& c) {
  return __builtin_amdgcn_wmma_f32_16x16x32_bf16(false, a, false, b, (short)0, c, false, false);
}

// Two contiguous 16-byte LDS runs -> one 16-element fragment (2x ds_load_b128).
static __device__ __forceinline__ v16bf ld_frag2(const __bf16* p0, const __bf16* p1) {
  bf16x8 lo = *(const bf16x8*)p0;
  bf16x8 hi = *(const bf16x8*)p1;
  return __builtin_shufflevector(lo, hi, 0, 1, 2, 3, 4, 5, 6, 7, 8, 9, 10, 11, 12, 13, 14, 15);
}

// Async global->LDS DMA of 16 bytes per lane (CDNA5 ASYNCcnt path).
static __device__ __forceinline__ void async_copy16(unsigned lds_off, const void* gsrc) {
  asm volatile("global_load_async_to_lds_b128 %0, %1, off"
               :: "v"(lds_off), "v"((unsigned long long)(size_t)gsrc)
               : "memory");
}
static __device__ __forceinline__ void async_wait0() {
  asm volatile("s_wait_asynccnt 0" ::: "memory");
}

// ---------------------------------------------------------------------------
// f32 -> bf16 conversion (weights), 8 elements per thread, b128 in / b128 out
// ---------------------------------------------------------------------------
__global__ __launch_bounds__(256)
void f32_to_bf16_kernel(const float* __restrict__ x, __bf16* __restrict__ y, int n)
{
  int i = (blockIdx.x * 256 + threadIdx.x) * 8;
  if (i >= n) return;
  f32x4 a = *(const f32x4*)(x + i);
  f32x4 b = *(const f32x4*)(x + i + 4);
  bf16x8 h;
#pragma unroll
  for (int j = 0; j < 4; ++j) { h[j] = (__bf16)a[j]; h[4 + j] = (__bf16)b[j]; }
  *(bf16x8*)(y + i) = h;
}

// ---------------------------------------------------------------------------
// LayerNorm: one block per row, D = 1024, 256 threads x 4 consecutive elems.
// bf16 output (feeds WMMA GEMMs).
// ---------------------------------------------------------------------------
__global__ __launch_bounds__(256)
void layernorm_kernel(const float* __restrict__ x, const float* __restrict__ g,
                      const float* __restrict__ b, __bf16* __restrict__ y)
{
  __shared__ float red[8];
  __shared__ float stat[2];
  const int row = blockIdx.x, tid = threadIdx.x, lane = tid & 31, w = tid >> 5;
  const float* xr = x + (size_t)row * CDIM;
  f32x4 lv = *(const f32x4*)(xr + tid * 4);
  float s = lv[0] + lv[1] + lv[2] + lv[3];
#pragma unroll
  for (int o = 16; o; o >>= 1) s += __shfl_xor(s, o, 32);
  if (!lane) red[w] = s;
  __syncthreads();
  if (tid == 0) { float t = 0.f; for (int i = 0; i < 8; ++i) t += red[i]; stat[0] = t / CDIM; }
  __syncthreads();
  const float mean = stat[0];
  float vs = 0.f;
#pragma unroll
  for (int i = 0; i < 4; ++i) { float d = lv[i] - mean; vs += d * d; }
#pragma unroll
  for (int o = 16; o; o >>= 1) vs += __shfl_xor(vs, o, 32);
  __syncthreads();
  if (!lane) red[w] = vs;
  __syncthreads();
  if (tid == 0) { float t = 0.f; for (int i = 0; i < 8; ++i) t += red[i];
                  stat[1] = rsqrtf(t / CDIM + 1e-5f); }
  __syncthreads();
  const float rs = stat[1];
  f32x4 gv = *(const f32x4*)(g + tid * 4);
  f32x4 bv = *(const f32x4*)(b + tid * 4);
  bf16x4 out;
#pragma unroll
  for (int i = 0; i < 4; ++i) out[i] = (__bf16)((lv[i] - mean) * rs * gv[i] + bv[i]);
  *(bf16x4*)(y + (size_t)row * CDIM + tid * 4) = out;
}

// ---------------------------------------------------------------------------
// bf16-WMMA GEMM:  C[M,N] = act( A[M,K] @ B[K,N] + bias ) + resid
// Block tile 128x128, 8 waves (4x2), wave tile 32x64, K step 32.
// A staged with async global->LDS DMA (pure bf16 copy); B staged transposed
// Bt[n][k] so operand fragments are 2x ds_load_b128.  OB16: bf16 output.
// ---------------------------------------------------------------------------
constexpr int GM_BM = 128, GM_BN = 128, GM_BK = 32;

template <bool HB, bool RL, bool HR, bool OB16>
__global__ __launch_bounds__(256)
void gemm_bf16_kernel(const __bf16* __restrict__ A, const __bf16* __restrict__ Bw,
                      void* __restrict__ Cd, const float* __restrict__ bias,
                      const float* __restrict__ resid, int M, int N, int K)
{
  __shared__ __bf16 As[GM_BM * GM_BK];   // 8 KB, row-major [m][k]
  __shared__ __bf16 Bt[GM_BN * GM_BK];   // 8 KB, transposed [n][k]
  const int tid = threadIdx.x, lane = tid & 31, w = tid >> 5;
  const int wm = w >> 1, wn = w & 1;
  const int bm = blockIdx.x * GM_BM, bn = blockIdx.y * GM_BN;
  const int m_ = lane & 15, hi = lane >> 4;

  const v8f vz = {0.f, 0.f, 0.f, 0.f, 0.f, 0.f, 0.f, 0.f};
  v8f acc[2][4];
#pragma unroll
  for (int i = 0; i < 2; ++i)
#pragma unroll
    for (int j = 0; j < 4; ++j) acc[i][j] = vz;

  // staging coordinates (16 consecutive elements per thread)
  const int fa = tid * 16;
  const int ar = fa >> 5, ac = fa & 31;       // A: 128 rows x 32 cols
  const int br = fa >> 7, bc = fa & 127;      // B: 32 rows x 128 cols
  const unsigned ldsA = (unsigned)(size_t)(As + fa);

  for (int k0 = 0; k0 < K; k0 += GM_BK) {
    {  // ---- stage A: two async 16B global->LDS DMAs per thread ----
      const __bf16* ga = A + (size_t)(bm + ar) * K + k0 + ac;
      async_copy16(ldsA, ga);
      async_copy16(ldsA + 16, ga + 8);
    }
    {  // ---- stage B transposed: b128 global reads, scattered b16 LDS ----
      const __bf16* gb = Bw + (size_t)(k0 + br) * N + bn + bc;
      bf16x8 t0 = *(const bf16x8*)gb;
      bf16x8 t1 = *(const bf16x8*)(gb + 8);
#pragma unroll
      for (int j = 0; j < 8; ++j) {
        Bt[(bc + j) * GM_BK + br] = t0[j];
        Bt[(bc + 8 + j) * GM_BK + br] = t1[j];
      }
    }
    if (k0 + GM_BK < K) {  // prefetch next tiles (global_prefetch_b8)
      __builtin_prefetch(A + (size_t)(bm + ar) * K + k0 + GM_BK + ac, 0, 1);
      __builtin_prefetch(Bw + (size_t)(k0 + GM_BK + br) * N + bn + bc, 0, 1);
    }
    async_wait0();
    __syncthreads();

#pragma unroll
    for (int mt = 0; mt < 2; ++mt) {
      const __bf16* ab = As + (wm * 32 + mt * 16 + m_) * GM_BK;
      v16bf af = ld_frag2(ab + hi * 8, ab + 16 + hi * 8);
#pragma unroll
      for (int nt = 0; nt < 4; ++nt) {
        const __bf16* bb = Bt + (wn * 64 + nt * 16 + m_) * GM_BK + hi * 8;
        v16bf bf = ld_frag2(bb, bb + 16);
        acc[mt][nt] = wmma_bf16(af, bf, acc[mt][nt]);
      }
    }
    __syncthreads();
  }

#pragma unroll
  for (int mt = 0; mt < 2; ++mt)
#pragma unroll
    for (int nt = 0; nt < 4; ++nt) {
      const int gn = bn + wn * 64 + nt * 16 + m_;
      const float bv = HB ? bias[gn] : 0.f;
#pragma unroll
      for (int r = 0; r < 8; ++r) {
        const int gm = bm + wm * 32 + mt * 16 + 8 * hi + r;
        float v = acc[mt][nt][r] + bv;
        if (RL) v = v > 0.f ? v : 0.f;
        if (HR) v += resid[(size_t)gm * N + gn];
        if (OB16) ((__bf16*)Cd)[(size_t)gm * N + gn] = (__bf16)v;
        else      ((float*)Cd)[(size_t)gm * N + gn] = v;
      }
    }
}

// ---------------------------------------------------------------------------
// LSH bucket assignment: one thread per (r,b,h,s), bf16 qk input.
// ---------------------------------------------------------------------------
__global__ __launch_bounds__(256)
void lsh_bucket_kernel(const __bf16* __restrict__ qk, const float* __restrict__ rot,
                       int* __restrict__ buckets)
{
  int idx = blockIdx.x * 256 + threadIdx.x;
  if (idx >= CR * CBATCH * CHEAD * CS) return;
  int s = idx % CS;
  int h = (idx / CS) % CHEAD;
  int b = (idx / (CS * CHEAD)) % CBATCH;
  int r = idx / (CS * CHEAD * CBATCH);
  const __bf16* q = qk + ((size_t)(b * CS + s)) * CDIM + h * CDH;
  float acc[NBH];
#pragma unroll
  for (int k = 0; k < NBH; ++k) acc[k] = 0.f;
  for (int d8 = 0; d8 < CDH; d8 += 8) {
    bf16x8 qv = *(const bf16x8*)(q + d8);
#pragma unroll
    for (int j = 0; j < 8; ++j) {
      float qd = (float)qv[j];
      const float* rp = rot + ((size_t)(d8 + j) * CR + r) * NBH;
#pragma unroll
      for (int k = 0; k < NBH; ++k) acc[k] += qd * rp[k];
    }
  }
  float best = -3.4e38f; int bi = 0;
#pragma unroll
  for (int k = 0; k < CNBUCK; ++k) {
    float v = (k < NBH) ? acc[k] : -acc[k - NBH];
    if (v > best) { best = v; bi = k; }   // strict > : first-max argmax semantics
  }
  buckets[idx] = bi;
}

// ---------------------------------------------------------------------------
// Stable counting sort by bucket per (r,b,h) row: order[dst] = src position.
// ---------------------------------------------------------------------------
constexpr int ST = 128, SEG = CS / ST;   // 32

__global__ __launch_bounds__(ST)
void bucket_sort_kernel(const int* __restrict__ buckets, int* __restrict__ order)
{
  __shared__ int cnt[CNBUCK][ST];        // 32 KB
  __shared__ int base[CNBUCK];
  const int row = blockIdx.x, tid = threadIdx.x;
  const int* bk = buckets + (size_t)row * CS;
  int* od = order + (size_t)row * CS;
  for (int bsel = 0; bsel < CNBUCK; ++bsel) cnt[bsel][tid] = 0;
  __syncthreads();
  const int start = tid * SEG;
  for (int i = 0; i < SEG; ++i) cnt[bk[start + i]][tid]++;
  __syncthreads();
  if (tid < CNBUCK) {                    // exclusive prefix over threads
    int run = 0;
    for (int t = 0; t < ST; ++t) { int c = cnt[tid][t]; cnt[tid][t] = run; run += c; }
    base[tid] = run;
  }
  __syncthreads();
  if (tid == 0) {                        // exclusive prefix over buckets
    int run = 0;
    for (int bb = 0; bb < CNBUCK; ++bb) { int t = base[bb]; base[bb] = run; run += t; }
  }
  __syncthreads();
  for (int i = 0; i < SEG; ++i) {
    int bb = bk[start + i];
    int pos = base[bb] + cnt[bb][tid];
    cnt[bb][tid]++;
    od[pos] = start + i;
  }
}

// ---------------------------------------------------------------------------
// Chunked LSH attention: one block per (r,b,h,chunk); 8 waves.
// Q gathered via async DMA; K L2-normalized; V transposed (Vt[d][key]).
// Score GEMM + masked softmax/logsumexp + output GEMM, all WMMA bf16.
// ---------------------------------------------------------------------------
__global__ __launch_bounds__(256)
void lsh_attn_kernel(const __bf16* __restrict__ qk, const __bf16* __restrict__ vv,
                     const int* __restrict__ buckets, const int* __restrict__ order,
                     float* __restrict__ o_all, float* __restrict__ l_all)
{
  __shared__ __bf16 Qs[CCH * CDH];     //  8 KB  [query][d]
  __shared__ __bf16 Ks[KE * CDH];      // 16 KB  [key][d]   (d-contiguous)
  __shared__ __bf16 Vt[CDH * KE];      // 16 KB  [d][key]   (key-contiguous)
  __shared__ __bf16 Ps[CCH * KE];      // 16 KB  [query][key]
  __shared__ int   pq[CCH], bq[CCH], pk[KE], bkk[KE];
  __shared__ float rmax[CCH], rsum[CCH];
  __shared__ float wred[CCH * 8];      //  2 KB

  const int idx = blockIdx.x;
  const int n = idx % CNCH;
  const int h = (idx / CNCH) % CHEAD;
  const int b = (idx / (CNCH * CHEAD)) % CBATCH;
  const int r = idx / (CNCH * CHEAD * CBATCH);
  const size_t rowbase = ((size_t)(r * CBATCH + b) * CHEAD + h);
  const int* od = order + rowbase * CS;
  const int* bk = buckets + rowbase * CS;

  const int tid = threadIdx.x, lane = tid & 31, w = tid >> 5;
  const int m_ = lane & 15, hi = lane >> 4;
  const int prev = (n + CNCH - 1) % CNCH;

  if (tid < CCH) {                                       // current chunk meta
    int orig = od[n * CCH + tid];
    pq[tid] = orig; bq[tid] = bk[orig];
    pk[tid] = orig; bkk[tid] = bq[tid];
  } else if (tid < 2 * CCH) {                            // previous chunk meta
    int j = tid - CCH;
    int orig = od[prev * CCH + j];
    pk[CCH + j] = orig; bkk[CCH + j] = bk[orig];
  }
  __syncthreads();

  {                                                      // Q gather: async DMA
    int ro = tid >> 2, c0 = (tid & 3) * 16;
    const __bf16* gq = qk + ((size_t)(b * CS + pq[ro])) * CDIM + h * CDH + c0;
    unsigned ldsQ = (unsigned)(size_t)(Qs + ro * CDH + c0);
    async_copy16(ldsQ, gq);
    async_copy16(ldsQ + 16, gq + 8);
  }
  {                                                      // K (L2-normed) + V^T
    int ro = tid >> 1, half = tid & 1;
    const __bf16* src = qk + ((size_t)(b * CS + pk[ro])) * CDIM + h * CDH + half * 32;
    bf16x8 kv[4];
#pragma unroll
    for (int q = 0; q < 4; ++q) kv[q] = *(const bf16x8*)(src + 8 * q);
    float ss = 0.f;
    float loc[32];
#pragma unroll
    for (int i = 0; i < 32; ++i) { loc[i] = (float)kv[i >> 3][i & 7]; ss += loc[i] * loc[i]; }
    ss += __shfl_xor(ss, 1, 32);
    float inv = 1.f / (sqrtf(ss) + 1e-6f);
    bf16x8 kq[4];
#pragma unroll
    for (int i = 0; i < 32; ++i) kq[i >> 3][i & 7] = (__bf16)(loc[i] * inv);
#pragma unroll
    for (int q = 0; q < 4; ++q) *(bf16x8*)(Ks + ro * CDH + half * 32 + 8 * q) = kq[q];

    const __bf16* sv = vv + ((size_t)(b * CS + pk[ro])) * CDIM + h * CDH + half * 32;
    bf16x8 vq[4];
#pragma unroll
    for (int q = 0; q < 4; ++q) vq[q] = *(const bf16x8*)(sv + 8 * q);
#pragma unroll
    for (int i = 0; i < 32; ++i) Vt[(half * 32 + i) * KE + ro] = vq[i >> 3][i & 7];
  }
  async_wait0();
  __syncthreads();

  // ---- score GEMM: wave w owns keys [w*16, w*16+16), all 64 query rows ----
  const v8f vz = {0.f, 0.f, 0.f, 0.f, 0.f, 0.f, 0.f, 0.f};
  v8f sc[4];
#pragma unroll
  for (int i = 0; i < 4; ++i) sc[i] = vz;

#pragma unroll
  for (int ks = 0; ks < CDH; ks += 32) {
    const __bf16* kb = Ks + (w * 16 + m_) * CDH + ks + hi * 8;  // B[d][key]=K[key][d]
    v16bf bfr = ld_frag2(kb, kb + 16);
#pragma unroll
    for (int mt = 0; mt < 4; ++mt) {
      const __bf16* qb = Qs + (mt * 16 + m_) * CDH + ks;
      v16bf afr = ld_frag2(qb + hi * 8, qb + 16 + hi * 8);
      sc[mt] = wmma_bf16(afr, bfr, sc[mt]);
    }
  }

  // ---- scale + bucket mask + self mask (C layout: n=lane&15, m=r+8*hi) ----
  const int keyj = w * 16 + m_;
  const int bkey = bkk[keyj], pkey = pk[keyj];
#pragma unroll
  for (int mt = 0; mt < 4; ++mt)
#pragma unroll
    for (int rr = 0; rr < 8; ++rr) {
      int qrow = mt * 16 + 8 * hi + rr;
      float sv_ = sc[mt][rr] * 0.125f;             // DH^-0.5
      if (bq[qrow] != bkey) sv_ = -1e9f;
      if (pq[qrow] == pkey) sv_ = -1e5f;
      sc[mt][rr] = sv_;
    }

  // ---- row max: 16-lane shfl reduction, then across waves in LDS ----
#pragma unroll
  for (int mt = 0; mt < 4; ++mt)
#pragma unroll
    for (int rr = 0; rr < 8; ++rr) {
      float v = sc[mt][rr];
#pragma unroll
      for (int o = 8; o; o >>= 1) v = fmaxf(v, __shfl_xor(v, o, 32));
      if (m_ == 0) wred[(mt * 16 + 8 * hi + rr) * 8 + w] = v;
    }
  __syncthreads();
  if (tid < CCH) {
    float v = wred[tid * 8];
    for (int i = 1; i < 8; ++i) v = fmaxf(v, wred[tid * 8 + i]);
    rmax[tid] = v;
  }
  __syncthreads();

  // ---- exp + row sum ----
#pragma unroll
  for (int mt = 0; mt < 4; ++mt)
#pragma unroll
    for (int rr = 0; rr < 8; ++rr) {
      int qrow = mt * 16 + 8 * hi + rr;
      float e = expf(sc[mt][rr] - rmax[qrow]);
      sc[mt][rr] = e;
      float v = e;
#pragma unroll
      for (int o = 8; o; o >>= 1) v += __shfl_xor(v, o, 32);
      if (m_ == 0) wred[qrow * 8 + w] = v;
    }
  __syncthreads();
  if (tid < CCH) {
    float s = 0.f;
    for (int i = 0; i < 8; ++i) s += wred[tid * 8 + i];
    rsum[tid] = s;
    l_all[rowbase * CS + pq[tid]] = rmax[tid] + logf(s);   // scatter logsumexp
  }
  __syncthreads();

  // ---- normalized probs -> LDS (bf16) ----
#pragma unroll
  for (int mt = 0; mt < 4; ++mt)
#pragma unroll
    for (int rr = 0; rr < 8; ++rr) {
      int qrow = mt * 16 + 8 * hi + rr;
      Ps[qrow * KE + keyj] = (__bf16)(sc[mt][rr] / rsum[qrow]);
    }
  __syncthreads();

  // ---- output GEMM: o(64x64) = P(64x128) @ V(128x64); 2 tiles per wave ----
  const int mt_o = w & 3, halfn = w >> 2;
  v8f oacc[2] = {vz, vz};
#pragma unroll
  for (int ks = 0; ks < KE; ks += 32) {
    const __bf16* pb = Ps + (mt_o * 16 + m_) * KE + ks;
    v16bf afr = ld_frag2(pb + hi * 8, pb + 16 + hi * 8);
#pragma unroll
    for (int u = 0; u < 2; ++u) {
      const __bf16* vb = Vt + ((halfn * 2 + u) * 16 + m_) * KE + ks + hi * 8;
      v16bf bfr = ld_frag2(vb, vb + 16);
      oacc[u] = wmma_bf16(afr, bfr, oacc[u]);
    }
  }
#pragma unroll
  for (int u = 0; u < 2; ++u)
#pragma unroll
    for (int rr = 0; rr < 8; ++rr) {
      int qrow = mt_o * 16 + 8 * hi + rr;
      int d = (halfn * 2 + u) * 16 + m_;
      o_all[(rowbase * CS + pq[qrow]) * CDH + d] = oacc[u][rr];
    }
}

// ---------------------------------------------------------------------------
// Combine rounds: softmax over R logsumexps; bf16 attn output (GEMM input).
// Each thread: 4 consecutive d.
// ---------------------------------------------------------------------------
__global__ __launch_bounds__(256)
void combine_kernel(const float* __restrict__ o_all, const float* __restrict__ l_all,
                    __bf16* __restrict__ attn)
{
  size_t i = ((size_t)blockIdx.x * 256 + threadIdx.x) * 4;
  if (i >= (size_t)CBATCH * CS * CDIM) return;
  int d = i % CDH;
  int h = (i / CDH) % CHEAD;
  int s = (i / CDIM) % CS;
  int b = i / ((size_t)CDIM * CS);
  float l[CR];
#pragma unroll
  for (int r = 0; r < CR; ++r)
    l[r] = l_all[((size_t)(r * CBATCH + b) * CHEAD + h) * CS + s];
  float mx = l[0];
#pragma unroll
  for (int r = 1; r < CR; ++r) mx = fmaxf(mx, l[r]);
  float Z = 0.f;
  f32x4 acc = {0.f, 0.f, 0.f, 0.f};
#pragma unroll
  for (int r = 0; r < CR; ++r) {
    float e = expf(l[r] - mx);
    Z += e;
    f32x4 ov = *(const f32x4*)(o_all +
        (((size_t)(r * CBATCH + b) * CHEAD + h) * CS + s) * CDH + d);
#pragma unroll
    for (int j = 0; j < 4; ++j) acc[j] += e * ov[j];
  }
  float iz = 1.f / Z;
  bf16x4 out;
#pragma unroll
  for (int j = 0; j < 4; ++j) out[j] = (__bf16)(acc[j] * iz);
  *(bf16x4*)(attn + i) = out;
}

// ---------------------------------------------------------------------------
// Host launcher
// ---------------------------------------------------------------------------
extern "C" void kernel_launch(void* const* d_in, const int* in_sizes, int n_in,
                              void* d_out, int out_size, void* d_ws, size_t ws_size,
                              hipStream_t stream) {
  const float* x1    = (const float*)d_in[0];
  const float* x2    = (const float*)d_in[1];
  const float* Wqk   = (const float*)d_in[2];
  const float* Wv    = (const float*)d_in[3];
  const float* Wo    = (const float*)d_in[4];
  const float* ln1_g = (const float*)d_in[5];
  const float* ln1_b = (const float*)d_in[6];
  const float* W1    = (const float*)d_in[7];
  const float* bf1   = (const float*)d_in[8];
  const float* W2    = (const float*)d_in[9];
  const float* bf2   = (const float*)d_in[10];
  const float* ln2_g = (const float*)d_in[11];
  const float* ln2_b = (const float*)d_in[12];
  const float* rot   = (const float*)d_in[13];

  const size_t BSD = (size_t)CBATCH * CS * CDIM;
  float* y1 = (float*)d_out;
  float* y2 = y1 + BSD;

  // ---- workspace layout (bf16 activations; OALL aliased with bf16 ffmid) ----
  char* ws = (char*)d_ws;
  size_t off = 0;
  __bf16* ACT  = (__bf16*)(ws + off); off += BSD * 2;               // 16.8 MB
  __bf16* QKb  = (__bf16*)(ws + off); off += BSD * 2;               // 16.8 MB
  __bf16* Vbb  = (__bf16*)(ws + off); off += BSD * 2;               // 16.8 MB
  __bf16* WQKb = (__bf16*)(ws + off); off += (size_t)CDIM * CDIM * 2;
  __bf16* WVb  = (__bf16*)(ws + off); off += (size_t)CDIM * CDIM * 2;
  __bf16* WOb  = (__bf16*)(ws + off); off += (size_t)CDIM * CDIM * 2;
  __bf16* W1b  = (__bf16*)(ws + off); off += (size_t)CDIM * CDFF * 2;
  __bf16* W2b  = (__bf16*)(ws + off); off += (size_t)CDFF * CDIM * 2;
  int*    BKT  = (int*)   (ws + off); off += (size_t)CR * CBATCH * CHEAD * CS * 4;
  int*    ORD  = (int*)   (ws + off); off += (size_t)CR * CBATCH * CHEAD * CS * 4;
  float*  LALL = (float*) (ws + off); off += (size_t)CR * CBATCH * CHEAD * CS * 4;
  float*  OALL = (float*) (ws + off);                               // 134 MB
  __bf16* FFM  = (__bf16*)OALL;       // alias (67 MB, used after OALL consumed)

  const int Mrows = CBATCH * CS;                       // 8192

  // 0) weights -> bf16 (deterministic, every call)
  auto cvt = [&](const float* src, __bf16* dst, size_t n) {
    f32_to_bf16_kernel<<<(unsigned)((n / 8 + 255) / 256), 256, 0, stream>>>(src, dst, (int)n);
  };
  cvt(Wqk, WQKb, (size_t)CDIM * CDIM);
  cvt(Wv,  WVb,  (size_t)CDIM * CDIM);
  cvt(Wo,  WOb,  (size_t)CDIM * CDIM);
  cvt(W1,  W1b,  (size_t)CDIM * CDFF);
  cvt(W2,  W2b,  (size_t)CDFF * CDIM);

  // 1) LN(x2) -> ACT (bf16)
  layernorm_kernel<<<Mrows, 256, 0, stream>>>(x2, ln1_g, ln1_b, ACT);
  // 2) qk / v projections (bf16 out)
  gemm_bf16_kernel<false, false, false, true>
      <<<dim3(Mrows / GM_BM, CDIM / GM_BN), 256, 0, stream>>>(
          ACT, WQKb, QKb, nullptr, nullptr, Mrows, CDIM, CDIM);
  gemm_bf16_kernel<false, false, false, true>
      <<<dim3(Mrows / GM_BM, CDIM / GM_BN), 256, 0, stream>>>(
          ACT, WVb, Vbb, nullptr, nullptr, Mrows, CDIM, CDIM);
  // 3) LSH buckets
  {
    int tot = CR * CBATCH * CHEAD * CS;
    lsh_bucket_kernel<<<(tot + 255) / 256, 256, 0, stream>>>(QKb, rot, BKT);
  }
  // 4) stable bucket sort -> order
  bucket_sort_kernel<<<CR * CBATCH * CHEAD, ST, 0, stream>>>(BKT, ORD);
  // 5) chunked attention (WMMA)
  lsh_attn_kernel<<<CR * CBATCH * CHEAD * CNCH, 256, 0, stream>>>(
      QKb, Vbb, BKT, ORD, OALL, LALL);
  // 6) combine rounds -> ACT (bf16)
  combine_kernel<<<(unsigned)((BSD / 4 + 255) / 256), 256, 0, stream>>>(OALL, LALL, ACT);
  // 7) y1 = x1 + attn @ Wo  (f32 out)
  gemm_bf16_kernel<false, false, true, false>
      <<<dim3(Mrows / GM_BM, CDIM / GM_BN), 256, 0, stream>>>(
          ACT, WOb, y1, nullptr, x1, Mrows, CDIM, CDIM);
  // 8) LN(y1) -> ACT (bf16)
  layernorm_kernel<<<Mrows, 256, 0, stream>>>(y1, ln2_g, ln2_b, ACT);
  // 9) ffmid = relu(LN(y1) @ W1 + b1)  (bf16 out)
  gemm_bf16_kernel<true, true, false, true>
      <<<dim3(Mrows / GM_BM, CDFF / GM_BN), 256, 0, stream>>>(
          ACT, W1b, FFM, bf1, nullptr, Mrows, CDFF, CDIM);
  // 10) y2 = x2 + ffmid @ W2 + b2  (f32 out)
  gemm_bf16_kernel<true, false, true, false>
      <<<dim3(Mrows / GM_BM, CDIM / GM_BN), 256, 0, stream>>>(
          FFM, W2b, y2, bf2, x2, Mrows, CDIM, CDFF);
}
